// Net_39230231281866
// MI455X (gfx1250) — matrix operand order repaired and checked by reading, not compiled, not run on hardware
//
#include <hip/hip_runtime.h>

// ---------- types for WMMA ----------
typedef __attribute__((ext_vector_type(16))) __bf16 v16bf;
typedef __attribute__((ext_vector_type(8)))  __bf16 v8bf;
typedef __attribute__((ext_vector_type(8)))  float  v8f;

__device__ __forceinline__ unsigned short f32_to_bf16(float f) {
    unsigned int u = __float_as_uint(f);
    unsigned int r = u + 0x7FFFu + ((u >> 16) & 1u);   // round-to-nearest-even
    return (unsigned short)(r >> 16);
}

// =====================================================================
// Kernel 1: per-image fused conv1 -> taylor sigmoid -> conv2 -> bf16 H2
//   x  : [8192,1,28,28] f32
//   H2 : [8192,4032]    bf16 (channel-major flatten: c*576 + y*24 + x)
// =====================================================================
__global__ void __launch_bounds__(256)
conv_fused(const float* __restrict__ x,
           const float* __restrict__ w1, const float* __restrict__ b1,
           const float* __restrict__ w2, const float* __restrict__ b2,
           unsigned short* __restrict__ h2) {
    __shared__ float xs[784];          // 28x28 input image
    __shared__ float s1[6 * 676];      // conv1 activations 6x26x26
    __shared__ float w1s[54], b1s[6], w2s[378], b2s[7];

    const int t   = threadIdx.x;
    const int img = blockIdx.x;

    const float* xi = x + (size_t)img * 784;
    for (int i = t; i < 784; i += 256) xs[i] = xi[i];
    if (t < 54) w1s[t] = w1[t];
    if (t < 6)  b1s[t] = b1[t];
    for (int i = t; i < 378; i += 256) w2s[i] = w2[i];
    if (t < 7)  b2s[t] = b2[t];
    __syncthreads();

    // conv1 (1->6, 3x3 valid) + Taylor sigmoid
    for (int o = t; o < 6 * 676; o += 256) {
        int c = o / 676, rem = o % 676;
        int y = rem / 26, xx = rem % 26;
        const float* wp = w1s + c * 9;
        float s = b1s[c];
#pragma unroll
        for (int i = 0; i < 3; ++i)
#pragma unroll
            for (int j = 0; j < 3; ++j)
                s = fmaf(xs[(y + i) * 28 + (xx + j)], wp[i * 3 + j], s);
        // 1 / (2 + t + t^2/2 + t^3/6 + t^4/24) with t = -s
        float tt  = -s;
        float den = 2.0f + tt * (1.0f + tt * (0.5f + tt * (1.0f / 6.0f + tt * (1.0f / 24.0f))));
        s1[o] = 1.0f / den;
    }
    __syncthreads();

    // conv2 (6->7, 3x3 valid), write flattened bf16 row
    unsigned short* h2i = h2 + (size_t)img * 4032;
    for (int o = t; o < 7 * 576; o += 256) {
        int c = o / 576, rem = o % 576;
        int y = rem / 24, xx = rem % 24;
        float s = b2s[c];
#pragma unroll
        for (int ic = 0; ic < 6; ++ic) {
            const float* wp = w2s + (c * 6 + ic) * 9;
            const float* sp = s1 + ic * 676 + y * 26 + xx;
#pragma unroll
            for (int i = 0; i < 3; ++i)
#pragma unroll
                for (int j = 0; j < 3; ++j)
                    s = fmaf(sp[i * 26 + j], wp[i * 3 + j], s);
        }
        h2i[o] = f32_to_bf16(s);
    }
}

// =====================================================================
// Kernel 2: fw1 f32 -> bf16 (row-major [128][4032])
// =====================================================================
__global__ void __launch_bounds__(256)
cvt_bf16(const float* __restrict__ in, unsigned short* __restrict__ out, int n) {
    int i = blockIdx.x * 256 + threadIdx.x;
    if (i < n) out[i] = f32_to_bf16(in[i]);
}

// =====================================================================
// Kernel 3: WMMA GEMM H2[8192,4032]bf16 x fw1^T -> sigmoid -> fc2 ->
//           log_softmax -> out[8192,10]
// Block: 256 threads = 8 waves = 4 row-groups x 2 K-halves.
// Each wave: 16 rows x 128 cols (8 tiles) over half of K (63 steps).
// Per k-step: issue A + all 8 B fragment loads into DISTINCT registers
// first, then the 8 WMMAs -> staggered s_wait_loadcnt, loads overlap MMA.
// =====================================================================
__global__ void __launch_bounds__(256)
fc_fused(const unsigned short* __restrict__ h2,    // [8192][4032] bf16
         const unsigned short* __restrict__ fw1b,  // [128][4032]  bf16
         const float* __restrict__ fb1,            // [128]
         const float* __restrict__ fw2,            // [10][128]
         const float* __restrict__ fb2,            // [10]
         float* __restrict__ out) {                // [8192][10]
    __shared__ float hs[64][132];     // partials, then sigmoid(fc1) tile
    __shared__ float fw2s[10 * 128];
    __shared__ float fb1s[128];
    __shared__ float fb2s[10];

    const int t        = threadIdx.x;
    const int wave     = t >> 5;
    const int lane     = t & 31;
    const int rowgroup = wave & 3;     // which 16-row strip
    const int khalf    = wave >> 2;    // which half of K
    const int mbase    = blockIdx.x * 64;

    for (int i = t; i < 1280; i += 256) fw2s[i] = fw2[i];
    if (t < 128) fb1s[t] = fb1[t];
    if (t < 10)  fb2s[t] = fb2[t];

    v8f acc[8];
#pragma unroll
    for (int j = 0; j < 8; ++j)
#pragma unroll
        for (int e = 0; e < 8; ++e) acc[j][e] = 0.0f;

    // A fragment addressing (16-bit A 16x32, this code's contiguous k order, same as B):
    //  lane<16  : row M=lane,    K = k+0..15
    //  lane>=16 : row M=lane-16, K = k+16..31
    const int arow = mbase + rowgroup * 16 + (lane & 15);
    const unsigned short* aptr = h2 + (size_t)arow * 4032 + ((lane & 16) ? 16 : 0);
    // B fragment addressing (16-bit B 32x16, this code's contiguous k order, same as A):
    //  lane<16  : col N=lane,    K = k+0..15   (contiguous in fw1 row)
    //  lane>=16 : col N=lane-16, K = k+16..31
    const unsigned short* bbase =
        fw1b + (size_t)(lane & 15) * 4032 + (lane & 16);

    const int k0 = khalf * 2016;       // 4032 / 2; 2016 = 63 * 32
    const int k1 = k0 + 2016;
    for (int k = k0; k < k1; k += 32) {
        // ---- issue ALL loads for this k-step first ----
        union { v16bf v; v8bf h[2]; } a;
        a.h[0] = *(const v8bf*)(const void*)(aptr + k);
        a.h[1] = *(const v8bf*)(const void*)(aptr + k + 8);
        v16bf bfrag[8];
#pragma unroll
        for (int j = 0; j < 8; ++j)
            bfrag[j] = *(const v16bf*)(const void*)(bbase + (size_t)j * (16 * 4032) + k);
        // ---- then consume: waits stagger, later loads overlap WMMA ----
#pragma unroll
        for (int j = 0; j < 8; ++j)
            acc[j] = __builtin_amdgcn_wmma_f32_16x16x32_bf16(
                false, a.v, false, bfrag[j], (short)0, acc[j], false, false);
    }

    // C/D layout: VGPR r -> M=r (lanes 0-15) / M=r+8 (lanes 16-31), N=lane&15
    const int rbase = rowgroup * 16 + ((lane & 16) ? 8 : 0);

    // Phase 1: K-half-1 waves park raw partial sums in LDS
    if (khalf == 1) {
#pragma unroll
        for (int j = 0; j < 8; ++j) {
            const int col = j * 16 + (lane & 15);
#pragma unroll
            for (int r = 0; r < 8; ++r)
                hs[rbase + r][col] = acc[j][r];
        }
    }
    __syncthreads();

    // Phase 2: K-half-0 waves combine, add bias, sigmoid, store
    if (khalf == 0) {
#pragma unroll
        for (int j = 0; j < 8; ++j) {
            const int col  = j * 16 + (lane & 15);
            const float bi = fb1s[col];
#pragma unroll
            for (int r = 0; r < 8; ++r) {
                float v = acc[j][r] + hs[rbase + r][col] + bi;
                hs[rbase + r][col] = 1.0f / (1.0f + __expf(-v));
            }
        }
    }
    __syncthreads();

    // fc2 (10x128) + log_softmax, one thread per row
    if (t < 64) {
        const int row = t;
        float lg[10];
#pragma unroll
        for (int o = 0; o < 10; ++o) {
            float s = fb2s[o];
            const float* wrow = fw2s + o * 128;
            for (int c = 0; c < 128; ++c) s = fmaf(hs[row][c], wrow[c], s);
            lg[o] = s;
        }
        float mx = lg[0];
#pragma unroll
        for (int o = 1; o < 10; ++o) mx = fmaxf(mx, lg[o]);
        float sum = 0.0f;
#pragma unroll
        for (int o = 0; o < 10; ++o) sum += __expf(lg[o] - mx);
        const float lse = mx + __logf(sum);
        float* op = out + (size_t)(mbase + row) * 10;
#pragma unroll
        for (int o = 0; o < 10; ++o) op[o] = lg[o] - lse;
    }
}

// =====================================================================
// Host launcher
// inputs: x, w1, b1, w2, b2, fw1, fb1, fw2, fb2  (all f32)
// =====================================================================
extern "C" void kernel_launch(void* const* d_in, const int* in_sizes, int n_in,
                              void* d_out, int out_size, void* d_ws, size_t ws_size,
                              hipStream_t stream) {
    (void)in_sizes; (void)n_in; (void)out_size; (void)ws_size;
    const float* x   = (const float*)d_in[0];
    const float* w1  = (const float*)d_in[1];
    const float* b1  = (const float*)d_in[2];
    const float* w2  = (const float*)d_in[3];
    const float* b2  = (const float*)d_in[4];
    const float* fw1 = (const float*)d_in[5];
    const float* fb1 = (const float*)d_in[6];
    const float* fw2 = (const float*)d_in[7];
    const float* fb2 = (const float*)d_in[8];

    const size_t B = 8192, KDIM = 4032, NDIM = 128;
    unsigned short* H2   = (unsigned short*)d_ws;                         // B*KDIM bf16
    unsigned short* FW1B = (unsigned short*)((char*)d_ws + B * KDIM * 2); // NDIM*KDIM bf16

    conv_fused<<<(int)B, 256, 0, stream>>>(x, w1, b1, w2, b2, H2);
    cvt_bf16<<<(int)((NDIM * KDIM + 255) / 256), 256, 0, stream>>>(fw1, FW1B, (int)(NDIM * KDIM));
    fc_fused<<<(int)(B / 64), 256, 0, stream>>>(H2, FW1B, fb1, fw2, fb2, (float*)d_out);
}
